// PropagationLayers_54133767798955
// MI455X (gfx1250) — compile-verified, matrix operand order checked by reading, not run on hardware
//
#include <hip/hip_runtime.h>
#include <hip/hip_bf16.h>

// ---------------------------------------------------------------------------
// Problem constants (reference: B=2, N=512, C=128, D_IN=272, D_H=256)
// ---------------------------------------------------------------------------
#define BATCH 2
#define NPTS  512
#define CIN   128
#define DH    256      // hidden width (2C)
#define KADJ  16       // adj feature width (C/8)
#define JBLK  64       // j-rows per tile
#define HP    264      // padded LDS row stride (halves) for h tile
#define W2P   264      // padded LDS row stride (halves) for W2^T
#define W1P   24       // padded LDS row stride (halves) for W1c^T

typedef unsigned short us16;
typedef __attribute__((ext_vector_type(16))) __bf16 v16bf;
typedef __attribute__((ext_vector_type(8)))  float  v8f;

union FragAB { v16bf v; uint4 q[2]; unsigned w[8]; us16 u[16]; };
union AccF   { v8f   v; float f[8]; };

__device__ __forceinline__ us16 f2bf_sw(float f) {
  unsigned x = __float_as_uint(f);
  x += 0x7FFFu + ((x >> 16) & 1u);   // round-to-nearest-even
  return (us16)(x >> 16);
}

// packed f32x2 -> bf16x2: builtin if present, else native v_cvt_pk_bf16_f32
__device__ __forceinline__ unsigned pk_bf16(float a, float b) {
#if __has_builtin(__builtin_amdgcn_cvt_pk_bf16_f32)
  auto r = __builtin_amdgcn_cvt_pk_bf16_f32(a, b);
  unsigned u;
  __builtin_memcpy(&u, &r, 4);
  return u;
#else
  unsigned r;
  asm("v_cvt_pk_bf16_f32 %0, %1, %2" : "=v"(r) : "v"(a), "v"(b));
  return r;
#endif
}

// hardware tanh (gfx1250 v_tanh_f32) when available
__device__ __forceinline__ float fast_tanh(float x) {
#if __has_builtin(__builtin_amdgcn_tanhf)
  return __builtin_amdgcn_tanhf(x);
#elif __has_builtin(__builtin_amdgcn_tanh_f32)
  return __builtin_amdgcn_tanh_f32(x);
#else
  return tanhf(x);
#endif
}

// ---------------------------------------------------------------------------
// Prep 1: BN scale/shift folding + W1c' (scaled, transposed, bf16)
// ---------------------------------------------------------------------------
__global__ __launch_bounds__(DH) void prep_scales(
    const float* __restrict__ W1,
    const float* __restrict__ g1, const float* __restrict__ b1,
    const float* __restrict__ m1, const float* __restrict__ v1,
    const float* __restrict__ g2, const float* __restrict__ b2,
    const float* __restrict__ m2, const float* __restrict__ v2,
    float* __restrict__ s1, float* __restrict__ t1,
    float* __restrict__ s2, float* __restrict__ t2,
    us16* __restrict__ W1cT) {
  const int h = threadIdx.x;
  const float S1 = g1[h] * rsqrtf(v1[h] + 1e-5f);
  const float S2 = g2[h] * rsqrtf(v2[h] + 1e-5f);
  s1[h] = S1; t1[h] = b1[h] - m1[h] * S1;
  s2[h] = S2; t2[h] = b2[h] - m2[h] * S2;
#pragma unroll
  for (int k = 0; k < KADJ; ++k)
    W1cT[h * KADJ + k] = f2bf_sw(W1[(2 * CIN + k) * DH + h] * S1);
}

// ---------------------------------------------------------------------------
// Prep 2: W2T[n][k] = bf16(W2[k][n] * s2[n])   (256x256, bf16)
// ---------------------------------------------------------------------------
__global__ __launch_bounds__(256) void prep_w2t(
    const float* __restrict__ W2, const float* __restrict__ s2,
    us16* __restrict__ W2T) {
  const int idx = blockIdx.x * 256 + threadIdx.x;
  const int n = idx >> 8, k = idx & 255;
  W2T[n * DH + k] = f2bf_sw(W2[k * DH + n] * s2[n]);
}

// ---------------------------------------------------------------------------
// Prep 3: Ab[row][h] = t1[h] + s1[h] * sum_c in[row][c]*W1[c][h]
//         Bb[row][h] =          s1[h] * sum_c in[row][c]*W1[128+c][h]
// ---------------------------------------------------------------------------
__global__ __launch_bounds__(DH) void prep_ab(
    const float* __restrict__ in, const float* __restrict__ W1,
    const float* __restrict__ s1, const float* __restrict__ t1,
    float* __restrict__ Ab, float* __restrict__ Bb) {
  __shared__ float xin[CIN];
  const int row = blockIdx.x, h = threadIdx.x;
  if (h < CIN) xin[h] = in[row * CIN + h];
  __syncthreads();
  float a = 0.f, bv = 0.f;
#pragma unroll 4
  for (int c = 0; c < CIN; ++c) {
    const float x = xin[c];
    a  = fmaf(x, W1[c * DH + h], a);
    bv = fmaf(x, W1[(CIN + c) * DH + h], bv);
  }
  Ab[row * DH + h] = t1[h] + s1[h] * a;
  Bb[row * DH + h] = s1[h] * bv;
}

// ---------------------------------------------------------------------------
// Main kernel: one workgroup per (b,i); 8 waves; 8 j-blocks of 64 rows.
// LDS: W2T (256xW2P bf16) | hTile (64xHP bf16) | W1cT (256xW1P bf16)
//      | abias[256] f32 | t2[256] f32 | outAcc[256] f32
// ---------------------------------------------------------------------------
#define SMEM_BYTES ((256 * W2P + JBLK * HP + 256 * W1P) * 2 + 3 * DH * 4)

__global__ __launch_bounds__(256) void propagation_main(
    const float* __restrict__ adj, const float* __restrict__ Ab,
    const float* __restrict__ Bb,  const float* __restrict__ t2g,
    const us16* __restrict__ W1cTg, const us16* __restrict__ W2Tg,
    float* __restrict__ out) {
  extern __shared__ char smem[];
  us16* w2t   = (us16*)smem;                 // 256*W2P halves
  us16* htile = w2t + 256 * W2P;             // 64*HP halves
  us16* w1ct  = htile + JBLK * HP;           // 256*W1P halves
  float* fbase  = (float*)(w1ct + 256 * W1P);
  float* abias  = fbase;                     // 256
  float* t2s    = fbase + DH;                // 256
  float* outAcc = fbase + 2 * DH;            // 256

  const int tid = threadIdx.x;
  const int lane = tid & 31, wave = tid >> 5;
  const int lane16 = lane & 15;
  const int hi = lane >> 4;                  // 0 for lanes 0-15, 1 for 16-31
  const int kb8 = hi << 3;                   // K-run base within 16
  const int rb = wave & 3;                   // row block (16 j rows)
  const int ch = wave >> 2;                  // column half (128 cols)

  const int bi = blockIdx.x;                 // b*512 + i
  const int b  = bi >> 9;
  const int i  = bi & (NPTS - 1);

  // --- cooperative LDS fill -------------------------------------------------
  {
    // W2T: one 256-halves row per thread, moved with the async-DMA engine
    // (global_load_async_to_lds_b128: 16B per lane per issue, ASYNCcnt).
#pragma unroll
    for (int q = 0; q < DH / 8; ++q) {
      unsigned lds_b = (unsigned)(size_t)(w2t + (size_t)tid * W2P + q * 8);
      unsigned long long gaddr =
          (unsigned long long)(const void*)(W2Tg + (size_t)tid * DH + q * 8);
      asm volatile("global_load_async_to_lds_b128 %0, %1, off"
                   :: "v"(lds_b), "v"(gaddr) : "memory");
    }
    // W1cT: 16 halves per row (small, plain copy)
    const uint4* s1c = (const uint4*)(W1cTg + (size_t)tid * KADJ);
    uint4* d1c = (uint4*)(w1ct + (size_t)tid * W1P);
    d1c[0] = s1c[0]; d1c[1] = s1c[1];
    abias[tid]  = Ab[(size_t)bi * DH + tid];
    t2s[tid]    = t2g[tid];
    outAcc[tid] = 0.f;
    asm volatile("s_wait_asynccnt 0" ::: "memory");
  }
  __syncthreads();

  float colsum[8];
#pragma unroll
  for (int t = 0; t < 8; ++t) colsum[t] = 0.f;

  const size_t adj_row0 = (((size_t)b * NPTS + i) * NPTS) * KADJ;

  for (int jb = 0; jb < NPTS / JBLK; ++jb) {
    const int j0 = jb * JBLK;

    // ---- GEMM1: h = adj(64x16, pad K->32) @ W1c'(32x256) + biases, ReLU ----
    FragAB a1;
    {
      const int ja = j0 + rb * 16 + lane16;
      const float* ap = adj + adj_row0 + (size_t)ja * KADJ + kb8;
      const float4 f0 = ((const float4*)ap)[0];
      const float4 f1 = ((const float4*)ap)[1];
      a1.w[0] = pk_bf16(f0.x, f0.y);
      a1.w[1] = pk_bf16(f0.z, f0.w);
      a1.w[2] = pk_bf16(f1.x, f1.y);
      a1.w[3] = pk_bf16(f1.z, f1.w);
      a1.w[4] = 0; a1.w[5] = 0; a1.w[6] = 0; a1.w[7] = 0;
    }
    // prefetch next j-block's adj rows into cache while we compute
    if (jb + 1 < NPTS / JBLK) {
      const int jn = j0 + JBLK + rb * 16 + lane16;
      __builtin_prefetch(adj + adj_row0 + (size_t)jn * KADJ + kb8, 0, 1);
    }
#pragma unroll
    for (int t = 0; t < 8; ++t) {
      const int n = (ch * 8 + t) * 16 + lane16;
      FragAB bf;
      bf.q[0] = *(const uint4*)(w1ct + (size_t)n * W1P + kb8);
      bf.w[4] = 0; bf.w[5] = 0; bf.w[6] = 0; bf.w[7] = 0;
      AccF c;
#pragma unroll
      for (int r = 0; r < 8; ++r) c.f[r] = 0.f;
      c.v = __builtin_amdgcn_wmma_f32_16x16x32_bf16(
          false, a1.v, false, bf.v, (short)0, c.v, false, false);
      const float abn = abias[n];
      float vals[8];
#pragma unroll
      for (int r = 0; r < 8; ++r) {
        const int jg = j0 + rb * 16 + r + (hi << 3);
        vals[r] =
            fmaxf(c.f[r] + abn + Bb[((size_t)(b << 9) + jg) * DH + n], 0.f);
      }
#pragma unroll
      for (int p = 0; p < 4; ++p) {
        const unsigned pk = pk_bf16(vals[2 * p], vals[2 * p + 1]);
        const int m0 = rb * 16 + 2 * p + (hi << 3);
        htile[(size_t)m0 * HP + n]       = (us16)(pk & 0xFFFFu);
        htile[(size_t)(m0 + 1) * HP + n] = (us16)(pk >> 16);
      }
    }
    __syncthreads();

    // ---- GEMM2: z = h(64x256) @ W2'(256x256); tanh; col-sum ----------------
    AccF acc2[8];
#pragma unroll
    for (int t = 0; t < 8; ++t)
#pragma unroll
      for (int r = 0; r < 8; ++r) acc2[t].f[r] = 0.f;

#pragma unroll
    for (int kk = 0; kk < DH / 32; ++kk) {
      FragAB a2;
      const us16* hp0 = htile + (size_t)(rb * 16 + lane16) * HP + kk * 32 + kb8;
      a2.q[0] = *(const uint4*)hp0;
      a2.q[1] = *(const uint4*)(hp0 + 16);
#pragma unroll
      for (int t = 0; t < 8; ++t) {
        const int n = (ch * 8 + t) * 16 + lane16;
        const us16* wp = w2t + (size_t)n * W2P + kk * 32 + kb8;
        FragAB bf;
        bf.q[0] = *(const uint4*)wp;
        bf.q[1] = *(const uint4*)(wp + 16);
        acc2[t].v = __builtin_amdgcn_wmma_f32_16x16x32_bf16(
            false, a2.v, false, bf.v, (short)0, acc2[t].v, false, false);
      }
    }
#pragma unroll
    for (int t = 0; t < 8; ++t) {
      const int n = (ch * 8 + t) * 16 + lane16;
      const float tt = t2s[n];
#pragma unroll
      for (int r = 0; r < 8; ++r)
        colsum[t] += fast_tanh(acc2[t].f[r] + tt);
    }
    __syncthreads();   // before next j-block overwrites htile
  }

  // ---- reduce: lanes (L, L+16) share a column; waves share via LDS atomics -
#pragma unroll
  for (int t = 0; t < 8; ++t) {
    float v = colsum[t];
    v += __shfl_xor(v, 16, 32);
    if (lane < 16) atomicAdd(&outAcc[(ch * 8 + t) * 16 + lane16], v);
  }
  __syncthreads();
  out[(size_t)bi * DH + tid] = outAcc[tid];
}

// ---------------------------------------------------------------------------
// Launch
// ---------------------------------------------------------------------------
extern "C" void kernel_launch(void* const* d_in, const int* in_sizes, int n_in,
                              void* d_out, int out_size, void* d_ws, size_t ws_size,
                              hipStream_t stream) {
  (void)in_sizes; (void)n_in; (void)out_size; (void)ws_size;
  const float* inputs = (const float*)d_in[0];
  const float* adj    = (const float*)d_in[1];
  const float* W1     = (const float*)d_in[2];
  const float* g1     = (const float*)d_in[3];
  const float* b1     = (const float*)d_in[4];
  const float* m1     = (const float*)d_in[5];
  const float* v1     = (const float*)d_in[6];
  const float* W2     = (const float*)d_in[7];
  const float* g2     = (const float*)d_in[8];
  const float* b2     = (const float*)d_in[9];
  const float* m2     = (const float*)d_in[10];
  const float* v2     = (const float*)d_in[11];
  float* out = (float*)d_out;

  // workspace layout
  char* ws = (char*)d_ws;
  float* s1 = (float*)(ws + 0);
  float* t1 = (float*)(ws + 1024);
  float* s2 = (float*)(ws + 2048);
  float* t2 = (float*)(ws + 3072);
  float* Ab = (float*)(ws + 4096);                          // 1 MB
  float* Bb = (float*)(ws + 4096 + 1048576);                // 1 MB
  us16*  W1cT = (us16*)(ws + 4096 + 2 * 1048576);           // 8 KB
  us16*  W2T  = (us16*)(ws + 4096 + 2 * 1048576 + 8192);    // 128 KB

  hipFuncSetAttribute((const void*)propagation_main,
                      hipFuncAttributeMaxDynamicSharedMemorySize, SMEM_BYTES);

  prep_scales<<<1, DH, 0, stream>>>(W1, g1, b1, m1, v1, g2, b2, m2, v2,
                                    s1, t1, s2, t2, W1cT);
  prep_w2t<<<DH * DH / 256, 256, 0, stream>>>(W2, s2, W2T);
  prep_ab<<<BATCH * NPTS, DH, 0, stream>>>(inputs, W1, s1, t1, Ab, Bb);
  propagation_main<<<BATCH * NPTS, 256, SMEM_BYTES, stream>>>(
      adj, Ab, Bb, t2, W1cT, W2T, out);
}
